// VoxelPointsSampler_6657199309191
// MI455X (gfx1250) — matrix operand order repaired and checked by reading, not compiled, not run on hardware
//
#include <hip/hip_runtime.h>

// ---- problem constants (from reference setup) ----
#define B_   2
#define T_   4
#define FF   8        // B_*T_ frames
#define N_   4096
#define M_   128
#define C_   5
#define NS   32       // num_sample
#define FEAT 99       // 3 + 32 + 64
#define VOX  0.4f
#define PCX  (-75.2f)
#define PCY  (-75.2f)
#define GAM  1.1f
#define R1SQ (0.8f*0.8f)
#define R2SQ (1.6f*1.6f)
#define K1   16
#define K2   32
#define ACTW 32       // LDS activation row stride (channels)
#define NTILES 12     // prebaked f16 B tiles

typedef __attribute__((ext_vector_type(16))) _Float16 v16h;
typedef __attribute__((ext_vector_type(8)))  float    v8f;

// B-tile ids in the prebaked table
#define T_R1L1   0
#define T_R1L2   1
#define T_R1L3_0 2
#define T_R1L3_1 3
#define T_R2L1_0 4
#define T_R2L1_1 5
#define T_R2L2_0 6
#define T_R2L2_1 7
#define T_R2L3_0 8   // ..11

static __device__ __forceinline__ unsigned ballot32(bool p) {
  return __builtin_amdgcn_ballot_w32(p);
}
static __device__ __forceinline__ void wbar() {
  __builtin_amdgcn_wave_barrier();
}
static __device__ __forceinline__ v8f wmma_f16(v16h a, v16h b, v8f c) {
  // D = A(16x32 f16) * B(32x16 f16) + C(16x16 f32)
  return __builtin_amdgcn_wmma_f32_16x16x32_f16(false, a, false, b, (short)0, c,
                                                false, false);
}
static __device__ __forceinline__ v8f zero8() {
  v8f z;
  #pragma unroll
  for (int q = 0; q < 8; ++q) z[q] = 0.f;
  return z;
}
// cross-half (lane ^ 16) max via ds_swizzle SWAPX16: bit15=0, xor=0x10, and=0x1F
static __device__ __forceinline__ float xor16_max(float x) {
  int o = __builtin_amdgcn_ds_swizzle(__float_as_int(x), 0x401F);
  return fmaxf(x, __int_as_float(o));
}

// Load A tile (16x32, rows = samples mtile*16.., K = input channels) from LDS acts.
// ISA A-layout: lane<16: slots 0..7 = K0..7, 8..15 = K16..23 of row (lane&15);
//               lane>=16: slots 0..7 = K8..15, 8..15 = K24..31 of same row.
static __device__ __forceinline__ v16h load_a(const _Float16* act, int mtile,
                                              int cin, int lane) {
  int row = mtile * 16 + (lane & 15);
  int kb0 = (lane < 16) ? 0 : 8;
  const _Float16* r = act + row * ACTW;
  v16h a;
  #pragma unroll
  for (int h = 0; h < 8; ++h) {
    int k = kb0 + h;
    a[h] = (k < cin) ? r[k] : (_Float16)0.f;
  }
  #pragma unroll
  for (int h = 0; h < 8; ++h) {
    int k = kb0 + 16 + h;
    a[8 + h] = (k < cin) ? r[k] : (_Float16)0.f;
  }
  return a;
}

// D tile (C/D layout: vgpr d -> row mtile*16 + d + (lane>=16?8:0), col lane&15)
// + bias + relu -> f16 activations in LDS.
static __device__ __forceinline__ void store_d(_Float16* act, v8f d,
                                               const float* __restrict__ bias,
                                               int mtile, int ntile, int lane) {
  int col = lane & 15;
  int ch  = ntile * 16 + col;
  float bv = bias[ch];
  int mbase = mtile * 16 + ((lane < 16) ? 0 : 8);
  #pragma unroll
  for (int v = 0; v < 8; ++v) {
    float x = d[v] + bv;
    x = x > 0.f ? x : 0.f;
    act[(mbase + v) * ACTW + ch] = (_Float16)x;
  }
}

// -------- kernel 0: pre-bake fp32 weights into WMMA B-tile layout (f16) --------
// B-layout: lane L holds col N = ntile*16 + (L&15); h=L>>4: slots 0..7 = K 8h..8h+7, 8..15 = K 16+8h..16+8h+7.
// Table entry t stored at wt[(t*32 + lane)*16 + n].
__global__ void k_prepw(const float* __restrict__ w00,
                        const float* __restrict__ w01,
                        const float* __restrict__ w02,
                        const float* __restrict__ w10,
                        const float* __restrict__ w11,
                        const float* __restrict__ w12,
                        _Float16* __restrict__ wt) {
  int tile = blockIdx.x;
  int lane = threadIdx.x;  // 32 threads
  const float* w;
  int ic, nt;
  switch (tile) {
    case 0:  w = w00; ic = 5;  nt = 0; break;
    case 1:  w = w01; ic = 16; nt = 0; break;
    case 2:  w = w02; ic = 16; nt = 0; break;
    case 3:  w = w02; ic = 16; nt = 1; break;
    case 4:  w = w10; ic = 5;  nt = 0; break;
    case 5:  w = w10; ic = 5;  nt = 1; break;
    case 6:  w = w11; ic = 32; nt = 0; break;
    case 7:  w = w11; ic = 32; nt = 1; break;
    case 8:  w = w12; ic = 32; nt = 0; break;
    case 9:  w = w12; ic = 32; nt = 1; break;
    case 10: w = w12; ic = 32; nt = 2; break;
    default: w = w12; ic = 32; nt = 3; break;
  }
  _Float16* o = wt + ((size_t)tile * 32 + lane) * 16;
  int col = nt * 16 + (lane & 15);
  int hb = (lane >> 4) * 8;
  for (int n = 0; n < 16; ++n) {
    int k = (n < 8) ? (hb + n) : (16 + hb + (n - 8));
    int kk = (k < ic) ? k : 0;        // branchless clamp
    float v = w[col * ic + kk];
    o[n] = (_Float16)((k < ic) ? v : 0.f);
  }
}

// ---------------- kernel 1: per-point voxel validity ----------------
__global__ void k_valid(const float* __restrict__ pts,
                        const float* __restrict__ rois,
                        int* __restrict__ valid) {
  __shared__ float qx[M_], qy[M_], rv[M_];
  int f = blockIdx.y;
  const float* P = pts + (size_t)f * N_ * C_;
  const float* R = rois + (size_t)f * M_ * 7;
  for (int m = threadIdx.x; m < M_; m += blockDim.x) {
    float bx = R[m * 7 + 0], by = R[m * 7 + 1];
    float dx = R[m * 7 + 3], dy = R[m * 7 + 4];
    qx[m] = floorf((bx - PCX) / VOX);
    qy[m] = floorf((by - PCY) / VOX);
    float hd = 0.5f * sqrtf(dx * dx + dy * dy);
    rv[m] = ceilf(hd * GAM / VOX);
  }
  __syncthreads();
  int n = blockIdx.x * blockDim.x + threadIdx.x;
  float cx = floorf((P[n * C_ + 0] - PCX) / VOX);
  float cy = floorf((P[n * C_ + 1] - PCY) / VOX);
  int v = 0;
  for (int m = 0; m < M_; ++m) {
    if (fabsf(qx[m] - cx) < rv[m] && fabsf(qy[m] - cy) < rv[m]) { v = 1; break; }
  }
  valid[f * N_ + n] = v;
}

// ------- kernel 2: top_k(mask,32) == first-32-true-then-false (ballot scan) -------
__global__ void k_sample(const float* __restrict__ pts,
                         const float* __restrict__ rois,
                         const int* __restrict__ valid,
                         int* __restrict__ sidx,
                         float* __restrict__ sp_out) {
  __shared__ int sT[4][NS];
  __shared__ int sF[4][NS];
  int wid = threadIdx.x >> 5, lane = threadIdx.x & 31;
  int box = blockIdx.x * 4 + wid;
  int f = box / M_, m = box % M_;
  const float* P = pts + (size_t)f * N_ * C_;
  const float* R = rois + (size_t)f * M_ * 7 + m * 7;
  const int*   V = valid + f * N_;
  float bx = R[0], by = R[1];
  float ddx = R[3], ddy = R[4];
  float rad = GAM * 0.5f * sqrtf(ddx * ddx + ddy * ddy);
  float r2 = rad * rad;
  unsigned lt = (1u << lane) - 1u;
  int tc = 0, fc = 0;
  for (int base = 0; base < N_; base += 32) {
    int n = base + lane;
    float dx = P[n * C_ + 0] - bx, dy = P[n * C_ + 1] - by;
    bool pred = (dx * dx + dy * dy <= r2) && (V[n] != 0);
    unsigned bal = ballot32(pred);
    int pT = tc + __popc(bal & lt);
    int pF = fc + __popc((~bal) & lt);
    if (pred && pT < NS) sT[wid][pT] = n;
    if (!pred && pF < NS) sF[wid][pF] = n;
    tc += __popc(bal);
    fc += __popc(~bal);
    if (tc >= NS) break;  // uniform
  }
  wbar();
  int tcc = tc < NS ? tc : NS;
  int j = (lane < tcc) ? sT[wid][lane] : sF[wid][lane - tcc];
  float mk = (lane < tcc) ? 1.f : 0.f;
  sidx[box * NS + lane] = j;
  int b = f / T_, t = f % T_;
  size_t o = ((size_t)((b * M_ + m) * T_ + t) * NS + lane) * 5;
  #pragma unroll
  for (int c = 0; c < 5; ++c) sp_out[o + c] = P[j * C_ + c] * mk;
}

// ------- kernel 3: fused ball-group + 3-layer MLP (WMMA f16->f32) + maxpool -------
__global__ void __launch_bounds__(128)
k_mlp(const float* __restrict__ pts, const int* __restrict__ valid,
      const v16h* __restrict__ wt,
      const float* __restrict__ b00, const float* __restrict__ b01,
      const float* __restrict__ b02, const float* __restrict__ b10,
      const float* __restrict__ b11, const float* __restrict__ b12,
      float* __restrict__ feats) {
  __shared__ int      idx2[4][K2];
  __shared__ int      idx1[4][K1];
  __shared__ _Float16 actA[4][32][ACTW];
  __shared__ _Float16 actB[4][32][ACTW];

  int wid = threadIdx.x >> 5, lane = threadIdx.x & 31;
  int p = blockIdx.x * 4 + wid;
  int f = p >> 12;           // / N_
  int i = p & (N_ - 1);
  const float* P = pts + (size_t)f * N_ * C_;
  const int*   V = valid + f * N_;
  float cx = P[i * C_ + 0], cy = P[i * C_ + 1], cz = P[i * C_ + 2];
  unsigned lt = (1u << lane) - 1u;

  // ---- ball grouping: first-K valid neighbors in index order, both radii ----
  int c1 = 0, c2 = 0;
  for (int base = 0; base < N_; base += 32) {
    int jn = base + lane;
    int nb = (base + 32 < N_) ? (base + 32 + lane) : jn;
    __builtin_prefetch(P + (size_t)nb * C_, 0, 0);      // global_prefetch_b8
    const float* q = P + jn * C_;
    float dx = q[0] - cx;
    float dy = q[1] - cy;
    float dz = q[2] - cz;
    float d2 = dx * dx + dy * dy + dz * dz;
    bool vj = V[jn] != 0;
    bool p2 = vj && (d2 < R2SQ);
    bool p1 = vj && (d2 < R1SQ);
    unsigned bal2 = ballot32(p2);
    unsigned bal1 = ballot32(p1);
    int q2 = c2 + __popc(bal2 & lt);
    int q1 = c1 + __popc(bal1 & lt);
    if (p2 && q2 < K2) idx2[wid][q2] = jn;
    if (p1 && q1 < K1) idx1[wid][q1] = jn;
    c2 += __popc(bal2);
    c1 += __popc(bal1);
    if (c2 >= K2 && c1 >= K1) break;  // uniform
  }
  wbar();
  int n2 = c2 < K2 ? c2 : K2;
  int n1 = c1 < K1 ? c1 : K1;
  int f2 = (n2 > 0) ? idx2[wid][0] : 0;
  int f1 = (n1 > 0) ? idx1[wid][0] : 0;
  if (lane >= n2) idx2[wid][lane] = f2;                 // fill-with-first / 0
  if (lane < K1 && lane >= n1) idx1[wid][lane] = f1;
  wbar();

  // build layer-1 A tile from group rows (K=5, zero padded)
  auto build_a1 = [&](const int* lst, int mtile) -> v16h {
    v16h a;
    #pragma unroll
    for (int h = 0; h < 16; ++h) a[h] = (_Float16)0.f;
    if (lane < 16) {                                    // K slots 0..7 hold ch 0..4
      int j = lst[mtile * 16 + lane];
      const float* q = P + j * C_;
      a[0] = (_Float16)(q[0] - cx);
      a[1] = (_Float16)(q[1] - cy);
      a[2] = (_Float16)(q[2] - cz);
      a[3] = (_Float16)q[3];
      a[4] = (_Float16)q[4];
    }
    return a;
  };
  // prebaked B tile: one v16h (32B) per lane -> two global_load_b128
  auto ldb = [&](int tile) -> v16h { return wt[tile * 32 + lane]; };

  float* fr = feats + (size_t)(f * N_ + i) * FEAT;
  if (lane == 0) { fr[0] = cx; fr[1] = cy; fr[2] = cz; }

  // ================= radius 2 branch: 32 samples, 5->32->32->64 =================
  {
    v16h a0 = build_a1(idx2[wid], 0);
    v16h a1 = build_a1(idx2[wid], 1);
    #pragma unroll
    for (int nt = 0; nt < 2; ++nt) {                    // layer1: K=5 -> 32 ch
      v16h bt = ldb(T_R2L1_0 + nt);
      v8f d0 = wmma_f16(a0, bt, zero8());
      v8f d1 = wmma_f16(a1, bt, zero8());
      store_d(&actA[wid][0][0], d0, b10, 0, nt, lane);
      store_d(&actA[wid][0][0], d1, b10, 1, nt, lane);
    }
    wbar();
    a0 = load_a(&actA[wid][0][0], 0, 32, lane);
    a1 = load_a(&actA[wid][0][0], 1, 32, lane);
    #pragma unroll
    for (int nt = 0; nt < 2; ++nt) {                    // layer2: 32 -> 32
      v16h bt = ldb(T_R2L2_0 + nt);
      v8f d0 = wmma_f16(a0, bt, zero8());
      v8f d1 = wmma_f16(a1, bt, zero8());
      store_d(&actB[wid][0][0], d0, b11, 0, nt, lane);
      store_d(&actB[wid][0][0], d1, b11, 1, nt, lane);
    }
    wbar();
    a0 = load_a(&actB[wid][0][0], 0, 32, lane);
    a1 = load_a(&actB[wid][0][0], 1, 32, lane);
    #pragma unroll
    for (int nt = 0; nt < 4; ++nt) {                    // layer3: 32 -> 64 + maxpool
      v16h bt = ldb(T_R2L3_0 + nt);
      v8f d0 = wmma_f16(a0, bt, zero8());
      v8f d1 = wmma_f16(a1, bt, zero8());
      float bv = b12[nt * 16 + (lane & 15)];
      float mx = 0.f;                                   // relu output >= 0
      #pragma unroll
      for (int v = 0; v < 8; ++v) {
        float x0 = d0[v] + bv; if (x0 > mx) mx = x0;
        float x1 = d1[v] + bv; if (x1 > mx) mx = x1;
      }
      float tot = xor16_max(mx);                        // ds_swizzle SWAPX16
      if (lane < 16) fr[3 + 32 + nt * 16 + lane] = tot;
    }
  }

  // ================= radius 1 branch: 16 samples, 5->16->16->32 =================
  {
    v16h a0 = build_a1(idx1[wid], 0);
    {                                                   // layer1: K=5 -> 16 ch
      v16h bt = ldb(T_R1L1);
      v8f d0 = wmma_f16(a0, bt, zero8());
      store_d(&actA[wid][0][0], d0, b00, 0, 0, lane);
    }
    wbar();
    a0 = load_a(&actA[wid][0][0], 0, 16, lane);
    {                                                   // layer2: 16 -> 16
      v16h bt = ldb(T_R1L2);
      v8f d0 = wmma_f16(a0, bt, zero8());
      store_d(&actB[wid][0][0], d0, b01, 0, 0, lane);
    }
    wbar();
    a0 = load_a(&actB[wid][0][0], 0, 16, lane);
    #pragma unroll
    for (int nt = 0; nt < 2; ++nt) {                    // layer3: 16 -> 32 + maxpool
      v16h bt = ldb(T_R1L3_0 + nt);
      v8f d0 = wmma_f16(a0, bt, zero8());
      float bv = b02[nt * 16 + (lane & 15)];
      float mx = 0.f;
      #pragma unroll
      for (int v = 0; v < 8; ++v) {
        float x0 = d0[v] + bv; if (x0 > mx) mx = x0;
      }
      float tot = xor16_max(mx);
      if (lane < 16) fr[3 + nt * 16 + lane] = tot;
    }
  }
}

// ---------------- kernel 4: gather features for sampled points ----------------
__global__ void k_gather(const float* __restrict__ feats,
                         const int* __restrict__ sidx,
                         float* __restrict__ pf_out) {
  long x = (long)blockIdx.x * blockDim.x + threadIdx.x;
  const long total = (long)B_ * M_ * T_ * NS * FEAT;
  if (x >= total) return;
  int c = (int)(x % FEAT);
  long r = x / FEAT;
  int s = (int)(r % NS); r /= NS;
  int t = (int)(r % T_); r /= T_;
  int m = (int)(r % M_); r /= M_;
  int b = (int)r;
  int f = b * T_ + t;
  int j = sidx[((f * M_ + m) * NS) + s];
  pf_out[x] = feats[((size_t)(f * N_ + j)) * FEAT + c];
}

extern "C" void kernel_launch(void* const* d_in, const int* in_sizes, int n_in,
                              void* d_out, int out_size, void* d_ws, size_t ws_size,
                              hipStream_t stream) {
  (void)in_sizes; (void)n_in; (void)out_size; (void)ws_size;
  const float* pts  = (const float*)d_in[0];
  const float* rois = (const float*)d_in[1];
  const float* w00 = (const float*)d_in[2];  const float* b00 = (const float*)d_in[3];
  const float* w01 = (const float*)d_in[4];  const float* b01 = (const float*)d_in[5];
  const float* w02 = (const float*)d_in[6];  const float* b02 = (const float*)d_in[7];
  const float* w10 = (const float*)d_in[8];  const float* b10 = (const float*)d_in[9];
  const float* w11 = (const float*)d_in[10]; const float* b11 = (const float*)d_in[11];
  const float* w12 = (const float*)d_in[12]; const float* b12 = (const float*)d_in[13];

  float* out = (float*)d_out;
  float* sp_out = out;                                     // (B,M,T*NS,5)
  float* pf_out = out + (size_t)B_ * M_ * T_ * NS * 5;     // (B,M,T*NS,99)

  char* wsb = (char*)d_ws;
  int*      valid  = (int*)wsb;                                     // FF*N_ ints
  int*      sidx   = (int*)(wsb + (size_t)FF * N_ * 4);             // FF*M_*NS ints
  _Float16* wtiles = (_Float16*)(wsb + (size_t)FF * N_ * 4
                                     + (size_t)FF * M_ * NS * 4);   // 12*32*16 f16 (32B aligned)
  float*    feats  = (float*)((char*)wtiles + (size_t)NTILES * 32 * 16 * 2);

  k_prepw <<<NTILES, 32, 0, stream>>>(w00, w01, w02, w10, w11, w12, wtiles);
  k_valid <<<dim3(N_ / 256, FF), 256, 0, stream>>>(pts, rois, valid);
  k_sample<<<FF * M_ / 4, 128, 0, stream>>>(pts, rois, valid, sidx, sp_out);
  k_mlp   <<<FF * N_ / 4, 128, 0, stream>>>(pts, valid, (const v16h*)wtiles,
                                            b00, b01, b02, b10, b11, b12, feats);
  long total = (long)B_ * M_ * T_ * NS * FEAT;
  k_gather<<<(unsigned)((total + 255) / 256), 256, 0, stream>>>(feats, sidx, pf_out);
}
